// QTAttB_43739946942756
// MI455X (gfx1250) — matrix operand order repaired and checked by reading, not compiled, not run on hardware
//
#include <hip/hip_runtime.h>

#define NH 8
#define D  32
#define CW 4   // waves per coarse workgroup

typedef __attribute__((ext_vector_type(8)))  float  v8f;
typedef __attribute__((ext_vector_type(16))) __bf16 v16bf;
typedef __attribute__((ext_vector_type(4)))  int    v4i;

union Frag {
  v16bf v;
  unsigned int u[8];
  unsigned short h[16];
};

__device__ __forceinline__ unsigned short f2bfu(float f) {
  unsigned int u = __float_as_uint(f);
  u += 0x7fffu + ((u >> 16) & 1u);          // round-to-nearest-even
  return (unsigned short)(u >> 16);
}

// --------- gfx1250 async global->LDS copy (guarded; falls back cleanly) ----
#if defined(__has_builtin)
#if __has_builtin(__builtin_amdgcn_global_load_async_to_lds_b128) && \
    __has_builtin(__builtin_amdgcn_s_wait_asynccnt)
#define HAVE_ASYNC_LDS 1
#endif
#endif
#ifndef HAVE_ASYNC_LDS
#define HAVE_ASYNC_LDS 0
#endif

#if HAVE_ASYNC_LDS
#define ASG __attribute__((address_space(1)))
#define ASL __attribute__((address_space(3)))
__device__ __forceinline__ void async_b128(const void* g, void* l) {
  // AS1 addresses == flat addresses numerically; AS3 offset == low 32 bits of
  // the generic LDS address (aperture lives entirely in the high 32 bits).
  unsigned long long ga = (unsigned long long)g;
  unsigned int       la = (unsigned int)(unsigned long long)l;
  __builtin_amdgcn_global_load_async_to_lds_b128((ASG v4i*)ga, (ASL v4i*)la, 0, 0);
}
__device__ __forceinline__ void async_wait() { __builtin_amdgcn_s_wait_asynccnt(0); }
#else
__device__ __forceinline__ void async_wait() {}
#endif

// wave-private LDS phase fence (CDNA5: LDS ops are in-order per wave)
__device__ __forceinline__ void wave_sync() {
#if defined(__has_builtin) && __has_builtin(__builtin_amdgcn_fence)
  __builtin_amdgcn_fence(__ATOMIC_ACQ_REL, "wavefront");
#else
  __syncthreads();
#endif
}

// ---------------------------------------------------------------------------
// Pack (b, c=nh*32, S) f32  ->  [b][nh][S][32] bf16 (head-contiguous rows)
// ---------------------------------------------------------------------------
__global__ __launch_bounds__(256) void pack_kernel(
    const float* __restrict__ in, unsigned short* __restrict__ out, int S)
{
  __shared__ float tile[32][33];
  const int sc = blockIdx.x, head = blockIdx.y, b = blockIdx.z;
  const int tx = threadIdx.x & 31, ty = threadIdx.x >> 5;
  const float* ip = in + ((size_t)b * 256 + head * 32) * S + (size_t)sc * 32;
  #pragma unroll
  for (int j = 0; j < 4; ++j) {
    int d = ty + j * 8;
    tile[d][tx] = ip[(size_t)d * S + tx];
  }
  __syncthreads();
  unsigned int* op =
      (unsigned int*)(out + (((size_t)b * NH + head) * S + (size_t)sc * 32) * D);
  #pragma unroll
  for (int rpt = 0; rpt < 2; ++rpt) {
    int dw = (int)threadIdx.x + rpt * 256;   // 512 dwords per block
    int s  = dw >> 4;
    int t  = dw & 15;
    unsigned int lo = f2bfu(tile[2 * t][s]);
    unsigned int hi = f2bfu(tile[2 * t + 1][s]);
    op[dw] = lo | (hi << 16);
  }
}

// ---------------------------------------------------------------------------
// Coarse level: flash attention over 1024 keys. 4 waves/WG share staged K/V
// chunks; each wave owns one 16-query tile. WMMA bf16 16x16x32 QK^T and PV.
// ---------------------------------------------------------------------------
__global__ __launch_bounds__(32 * CW) void coarse_kernel(
    const unsigned short* __restrict__ pq,
    const unsigned short* __restrict__ pk,
    const unsigned short* __restrict__ pv,
    float* __restrict__ msg,   // [b][1024][nh][32]
    int*   __restrict__ tidx)  // [b][1024][16][nh]
{
  const int tid  = threadIdx.x;
  const int lane = tid & 31;
  const int wv   = tid >> 5;
  const int m    = lane & 15;
  const int hi   = lane >> 4;
  const int qt   = blockIdx.x * CW + wv;   // 64 query tiles / (b,head)
  const int head = blockIdx.y;
  const int b    = blockIdx.z;
  const int S    = 1024;

  __shared__ unsigned short Kbuf[32][32];       // shared chunk keys [key][d]
  __shared__ unsigned short VT[32][32];         // shared chunk values [d][key]
  __shared__ float sc[CW][16][32];              // per-wave scores
  __shared__ float rowmax[CW][16], rowsum[CW][16], alpha_s[CW][16];
  __shared__ float tvs[CW][16][32];
  __shared__ int   tis[CW][16][32];

  const size_t hb = ((size_t)b * NH + head) * S * D;

  // Q fragment (A layout 16x32 bf16)
  Frag qa;
  {
    const unsigned int* q32 =
        (const unsigned int*)(pq + hb + (size_t)(qt * 16 + m) * D);
    #pragma unroll
    for (int j = 0; j < 4; ++j) {
      qa.u[j]     = q32[hi * 8 + j];       // elems 0..7  : d = hi*16 + 0..7
      qa.u[4 + j] = q32[hi * 8 + 4 + j];   // elems 8..15 : d = hi*16 + 8..15
    }
  }

  if (lane < 16) { rowmax[wv][lane] = -1e30f; rowsum[wv][lane] = 0.0f; }
  v8f O0 = {}, O1 = {};
  float tv[16]; int ti[16];
  #pragma unroll
  for (int j = 0; j < 16; ++j) { tv[j] = -1e30f; ti[j] = 0; }
  float minv = -1e30f; int mslot = 0;

  for (int ck = 0; ck < 32; ++ck) {        // 32 chunks x 32 keys
    // ---- stage chunk: wave0 -> K (async DMA), wave1 -> V transposed ----
    if (wv == 0) {
#if HAVE_ASYNC_LDS
      const char* gs = (const char*)(pk + hb + (size_t)ck * 32 * D);  // 2KB contiguous
      char* ld = (char*)Kbuf;
      #pragma unroll
      for (int i = 0; i < 4; ++i)
        async_b128(gs + lane * 16 + i * 512, ld + lane * 16 + i * 512);
#else
      const unsigned int* kr =
          (const unsigned int*)(pk + hb + (size_t)(ck * 32 + lane) * D);
      unsigned int* kd = (unsigned int*)Kbuf[lane];
      #pragma unroll
      for (int j = 0; j < 8; ++j) kd[j] = kr[j];
#endif
    } else if (wv == 1) {
      const unsigned int* vr =
          (const unsigned int*)(pv + hb + (size_t)(ck * 32 + lane) * D);
      #pragma unroll
      for (int j = 0; j < 8; ++j) {
        unsigned int x = vr[j];
        VT[2 * j][lane]     = (unsigned short)(x & 0xffffu);
        VT[2 * j + 1][lane] = (unsigned short)(x >> 16);
      }
    }
    async_wait();
    __syncthreads();

    // ---- QK^T : two 16-key tiles ----
    #pragma unroll
    for (int t = 0; t < 2; ++t) {
      Frag kb;  // B frag: column n = key (t*16+m), K-range d = hi*16..+15
      const unsigned int* kr = (const unsigned int*)Kbuf[t * 16 + m];
      #pragma unroll
      for (int j = 0; j < 8; ++j) kb.u[j] = kr[hi * 8 + j];
      v8f cz = {};
      v8f s = __builtin_amdgcn_wmma_f32_16x16x32_bf16(false, qa.v, false, kb.v,
                                                      (short)0, cz, false, false);
      #pragma unroll
      for (int r = 0; r < 8; ++r)
        sc[wv][r + 8 * hi][t * 16 + m] = s[r] * 0.17677669529663687f; // 1/sqrt(32)
    }
    wave_sync();

    // ---- streaming top-16 (lane pair (m, m+16) owns half of row m) ----
    float cmax = -1e30f;
    #pragma unroll 1
    for (int j = 0; j < 16; ++j) {
      float v = sc[wv][m][hi * 16 + j];
      cmax = fmaxf(cmax, v);
      if (v > minv) {
        int gi = ck * 32 + hi * 16 + j;
        #pragma unroll
        for (int q = 0; q < 16; ++q) {
          bool rep = (q == mslot);
          tv[q] = rep ? v : tv[q];
          ti[q] = rep ? gi : ti[q];
        }
        minv = tv[0]; mslot = 0;
        #pragma unroll
        for (int q = 1; q < 16; ++q) {
          bool lt = tv[q] < minv;
          minv  = lt ? tv[q] : minv;
          mslot = lt ? q : mslot;
        }
      }
    }
    cmax = fmaxf(cmax, __shfl_xor(cmax, 16, 32));
    if (hi == 0) {
      float mold = rowmax[wv][m];
      float mnew = fmaxf(mold, cmax);
      rowmax[wv][m]  = mnew;
      alpha_s[wv][m] = __expf(mold - mnew);
    }
    wave_sync();

    // ---- exp once; write exp'd scores back; accumulate row sums ----
    const float mrow = rowmax[wv][m];
    float esum = 0.0f;
    #pragma unroll
    for (int j = 0; j < 16; ++j) {
      float e = __expf(sc[wv][m][hi * 16 + j] - mrow);
      esum += e;
      sc[wv][m][hi * 16 + j] = e;
    }
    esum += __shfl_xor(esum, 16, 32);
    if (hi == 0) rowsum[wv][m] = rowsum[wv][m] * alpha_s[wv][m] + esum;

    // rescale accumulators (element r of C <-> row r + 8*hi)
    #pragma unroll
    for (int r = 0; r < 8; ++r) {
      float a = alpha_s[wv][r + 8 * hi];
      O0[r] *= a; O1[r] *= a;
    }
    wave_sync();

    // ---- P fragment (A layout) from exp'd scores; V B-frags from VT ----
    Frag pf;
    #pragma unroll
    for (int j = 0; j < 8; ++j) {
      pf.h[j]     = f2bfu(sc[wv][m][hi * 16 + j]);
      pf.h[8 + j] = f2bfu(sc[wv][m][hi * 16 + 8 + j]);
    }
    Frag vb0, vb1;
    {
      const unsigned int* v0r = (const unsigned int*)VT[m];
      const unsigned int* v1r = (const unsigned int*)VT[16 + m];
      #pragma unroll
      for (int j = 0; j < 8; ++j) { vb0.u[j] = v0r[hi * 8 + j]; vb1.u[j] = v1r[hi * 8 + j]; }
    }
    O0 = __builtin_amdgcn_wmma_f32_16x16x32_bf16(false, pf.v, false, vb0.v, (short)0, O0, false, false);
    O1 = __builtin_amdgcn_wmma_f32_16x16x32_bf16(false, pf.v, false, vb1.v, (short)0, O1, false, false);
    __syncthreads();   // protect Kbuf/VT before next stage
  }
  wave_sync();

  // ---- write message (b, L, nh, d) ----
  #pragma unroll
  for (int r = 0; r < 8; ++r) {
    int row = r + 8 * hi;
    float inv = 1.0f / rowsum[wv][row];
    float* mo = msg + (((size_t)b * S + (qt * 16 + row)) * NH + head) * D;
    mo[m]      = O0[r] * inv;
    mo[16 + m] = O1[r] * inv;
  }

  // ---- merge lane-pair top-16 lists -> global top-16 (stable) ----
  #pragma unroll
  for (int j = 0; j < 16; ++j) { tvs[wv][m][hi * 16 + j] = tv[j]; tis[wv][m][hi * 16 + j] = ti[j]; }
  wave_sync();
  if (hi == 0) {
    int* to = tidx + (((size_t)b * S + (qt * 16 + m)) * 16) * NH + head;
    #pragma unroll 1
    for (int k = 0; k < 16; ++k) {
      float bv = -1e31f; int bj = 0; int bg = 0x7fffffff;
      #pragma unroll 1
      for (int j = 0; j < 32; ++j) {
        float v = tvs[wv][m][j];
        int   g = tis[wv][m][j];
        if (v > bv || (v == bv && g < bg)) { bv = v; bj = j; bg = g; }
      }
      to[k * NH] = bg;
      tvs[wv][m][bj] = -1e32f;
    }
  }
}

// ---------------------------------------------------------------------------
// Fine level: per (b, l, head): 4 queries (2x2 group) vs K4 gathered keys.
// ---------------------------------------------------------------------------
template <int K4, int KPREV, bool DO_TOPK>
__global__ __launch_bounds__(32) void fine_kernel(
    const unsigned short* __restrict__ pq,   // [b][nh][w*w][32]
    const unsigned short* __restrict__ pk,
    const unsigned short* __restrict__ pv,
    const int* __restrict__ pidx,            // [b][L][KPREV][nh]
    float* __restrict__ msg,                 // [b][L][4][nh][32]
    int*   __restrict__ tout,                // [b][w*w][8][nh]
    int w)
{
  const int lane = threadIdx.x;
  const int m    = lane & 15;
  const int hi   = lane >> 4;
  const int l    = blockIdx.x;
  const int head = blockIdx.y;
  const int b    = blockIdx.z;
  const int wp   = w >> 1;
  const int L    = wp * wp;
  const int r0 = l / wp, c0 = l % wp;

  __shared__ int   gidx[K4];
  __shared__ unsigned short Kbuf[K4][32];
  __shared__ unsigned short VT[32][K4];
  __shared__ float sc[4][K4];
  __shared__ float sck[4][K4];
  __shared__ float smax[4], ssum[4];

  if (lane < KPREV) {
    int p  = pidx[(((size_t)b * L + l) * KPREV + lane) * NH + head];
    int pr = (p / wp) * 2, pc = (p % wp) * 2;
    gidx[lane * 4 + 0] = pr * w + pc;
    gidx[lane * 4 + 1] = pr * w + pc + 1;
    gidx[lane * 4 + 2] = (pr + 1) * w + pc;
    gidx[lane * 4 + 3] = (pr + 1) * w + pc + 1;
  }
  __syncthreads();

  const size_t hb = ((size_t)b * NH + head) * (size_t)(w * w) * D;

  // gather K rows (async DMA when available) and V rows (transposed)
  for (int j = lane; j < K4; j += 32) {
#if HAVE_ASYNC_LDS
    const char* gs = (const char*)(pk + hb + (size_t)gidx[j] * D);
    char* ld = (char*)Kbuf[j];
    #pragma unroll
    for (int i = 0; i < 4; ++i) async_b128(gs + i * 16, ld + i * 16);
#else
    const unsigned int* kr = (const unsigned int*)(pk + hb + (size_t)gidx[j] * D);
    unsigned int* kd = (unsigned int*)Kbuf[j];
    #pragma unroll
    for (int q = 0; q < 8; ++q) kd[q] = kr[q];
#endif
    const unsigned int* vr = (const unsigned int*)(pv + hb + (size_t)gidx[j] * D);
    #pragma unroll
    for (int q = 0; q < 8; ++q) {
      unsigned int x = vr[q];
      VT[2 * q][j]     = (unsigned short)(x & 0xffffu);
      VT[2 * q + 1][j] = (unsigned short)(x >> 16);
    }
  }

  // Q fragment: rows 0..3 = 2x2 query group
  Frag qa;
  if (m < 4) {
    int pos = (2 * r0 + (m >> 1)) * w + (2 * c0 + (m & 1));
    const unsigned int* q32 = (const unsigned int*)(pq + hb + (size_t)pos * D);
    #pragma unroll
    for (int j = 0; j < 4; ++j) {
      qa.u[j]     = q32[hi * 8 + j];
      qa.u[4 + j] = q32[hi * 8 + 4 + j];
    }
  } else {
    #pragma unroll
    for (int j = 0; j < 8; ++j) qa.u[j] = 0u;
  }
  async_wait();
  __syncthreads();

  // QK^T over K4 keys
  #pragma unroll
  for (int t = 0; t < K4 / 16; ++t) {
    Frag kb;
    const unsigned int* kr = (const unsigned int*)Kbuf[t * 16 + m];
    #pragma unroll
    for (int j = 0; j < 8; ++j) kb.u[j] = kr[hi * 8 + j];
    v8f cz = {};
    v8f s = __builtin_amdgcn_wmma_f32_16x16x32_bf16(false, qa.v, false, kb.v,
                                                    (short)0, cz, false, false);
    if (hi == 0) {
      #pragma unroll
      for (int r = 0; r < 4; ++r) sc[r][t * 16 + m] = s[r] * 0.17677669529663687f;
    }
  }
  __syncthreads();

  // softmax stats per query (lanes 0..3); keep raw copy for top-k, exp in place
  if (lane < 4) {
    float mx = -1e30f;
    for (int j = 0; j < K4; ++j) mx = fmaxf(mx, sc[lane][j]);
    float sm = 0.0f;
    for (int j = 0; j < K4; ++j) {
      float e = __expf(sc[lane][j] - mx);
      sm += e;
      sck[lane][j] = sc[lane][j];
      sc[lane][j] = e;
    }
    smax[lane] = mx; ssum[lane] = sm;
  }
  __syncthreads();

  // top-8 over gathered keys (raw scores), mapped back to global indices
  if (DO_TOPK && lane < 4) {
    int pos = (2 * r0 + (lane >> 1)) * w + (2 * c0 + (lane & 1));
    int* to = tout + (((size_t)b * (w * w) + pos) * 8) * NH + head;
    #pragma unroll 1
    for (int k = 0; k < 8; ++k) {
      float bv = -1e31f; int bj = 0;
      #pragma unroll 1
      for (int j = 0; j < K4; ++j) {
        float v = sck[lane][j];
        if (v > bv) { bv = v; bj = j; }
      }
      to[k * NH] = gidx[bj];
      sck[lane][bj] = -1e32f;
    }
  }

  // PV: accumulate in 32-key chunks
  v8f O0 = {}, O1 = {};
  #pragma unroll
  for (int ck = 0; ck < K4 / 32; ++ck) {
    Frag pf;
    if (m < 4) {
      #pragma unroll
      for (int j = 0; j < 8; ++j) {
        pf.h[j]     = f2bfu(sc[m][ck * 32 + hi * 16 + j]);
        pf.h[8 + j] = f2bfu(sc[m][ck * 32 + hi * 16 + 8 + j]);
      }
    } else {
      #pragma unroll
      for (int j = 0; j < 8; ++j) pf.u[j] = 0u;
    }
    Frag vb0, vb1;
    const unsigned int* v0r = (const unsigned int*)VT[m];
    const unsigned int* v1r = (const unsigned int*)VT[16 + m];
    #pragma unroll
    for (int j = 0; j < 8; ++j) {
      vb0.u[j] = v0r[ck * 16 + hi * 8 + j];
      vb1.u[j] = v1r[ck * 16 + hi * 8 + j];
    }
    O0 = __builtin_amdgcn_wmma_f32_16x16x32_bf16(false, pf.v, false, vb0.v, (short)0, O0, false, false);
    O1 = __builtin_amdgcn_wmma_f32_16x16x32_bf16(false, pf.v, false, vb1.v, (short)0, O1, false, false);
  }

  if (hi == 0) {
    #pragma unroll
    for (int r = 0; r < 4; ++r) {
      float inv = 1.0f / ssum[r];
      float* mo = msg + ((((size_t)b * L + l) * 4 + r) * NH + head) * D;
      mo[m]      = O0[r] * inv;
      mo[16 + m] = O1[r] * inv;
    }
  }
}

// ---------------------------------------------------------------------------
// Hierarchical combine with softmax(weight)
// ---------------------------------------------------------------------------
__global__ __launch_bounds__(256) void combine_kernel(
    const float* __restrict__ m0,   // [b][1024][8][32]
    const float* __restrict__ m1,   // [b][1024][4][8][32]
    const float* __restrict__ m2,   // [b][4096][4][8][32]
    const float* __restrict__ wt,   // [3]
    float* __restrict__ out)        // [b][16384][8][32]
{
  size_t tid = (size_t)blockIdx.x * blockDim.x + threadIdx.x;
  if (tid >= (size_t)4 * 16384 * NH * D) return;
  int dd = (int)(tid & 31);
  int n  = (int)((tid >> 5) & 7);
  int p  = (int)((tid >> 8) & 16383);
  int b  = (int)(tid >> 22);
  int r = p >> 7, c = p & 127;

  float w0 = wt[0], w1 = wt[1], w2 = wt[2];
  float mx = fmaxf(w0, fmaxf(w1, w2));
  float e0 = __expf(w0 - mx), e1 = __expf(w1 - mx), e2 = __expf(w2 - mx);
  float inv = 1.0f / (e0 + e1 + e2);
  e0 *= inv; e1 *= inv; e2 *= inv;

  int l2 = (r >> 1) * 64 + (c >> 1);
  int t2 = (r & 1) * 2 + (c & 1);
  int l1 = (r >> 2) * 32 + (c >> 2);
  int t1 = ((r >> 1) & 1) * 2 + ((c >> 1) & 1);

  float v = e0 * m0[(((size_t)b * 1024 + l1) * NH + n) * D + dd]
          + e1 * m1[((((size_t)b * 1024 + l1) * 4 + t1) * NH + n) * D + dd]
          + e2 * m2[((((size_t)b * 4096 + l2) * 4 + t2) * NH + n) * D + dd];
  out[tid] = v;
}

// ---------------------------------------------------------------------------
extern "C" void kernel_launch(void* const* d_in, const int* in_sizes, int n_in,
                              void* d_out, int out_size, void* d_ws, size_t ws_size,
                              hipStream_t stream)
{
  (void)in_sizes; (void)n_in; (void)out_size; (void)ws_size;
  const float* q0 = (const float*)d_in[0];
  const float* q1 = (const float*)d_in[1];
  const float* q2 = (const float*)d_in[2];
  const float* k0 = (const float*)d_in[3];
  const float* k1 = (const float*)d_in[4];
  const float* k2 = (const float*)d_in[5];
  const float* v0 = (const float*)d_in[6];
  const float* v1 = (const float*)d_in[7];
  const float* v2 = (const float*)d_in[8];
  const float* wt = (const float*)d_in[9];

  const int S0 = 16384, S1 = 4096, S2 = 1024;
  size_t off = 0;
  char* ws = (char*)d_ws;
  auto alloc = [&](size_t bytes) -> void* {
    void* p = ws + off;
    off += (bytes + 255) & ~(size_t)255;
    return p;
  };
  unsigned short* pq0 = (unsigned short*)alloc((size_t)2 * 4 * NH * S0 * D);
  unsigned short* pq1 = (unsigned short*)alloc((size_t)2 * 4 * NH * S1 * D);
  unsigned short* pq2 = (unsigned short*)alloc((size_t)2 * 4 * NH * S2 * D);
  unsigned short* pk0 = (unsigned short*)alloc((size_t)2 * 4 * NH * S0 * D);
  unsigned short* pk1 = (unsigned short*)alloc((size_t)2 * 4 * NH * S1 * D);
  unsigned short* pk2 = (unsigned short*)alloc((size_t)2 * 4 * NH * S2 * D);
  unsigned short* pv0 = (unsigned short*)alloc((size_t)2 * 4 * NH * S0 * D);
  unsigned short* pv1 = (unsigned short*)alloc((size_t)2 * 4 * NH * S1 * D);
  unsigned short* pv2 = (unsigned short*)alloc((size_t)2 * 4 * NH * S2 * D);
  float* m0buf = (float*)alloc((size_t)4 * 4 * 1024 * NH * D);
  float* m1buf = (float*)alloc((size_t)4 * 4 * 1024 * 4 * NH * D);
  float* m2buf = (float*)alloc((size_t)4 * 4 * 4096 * 4 * NH * D);
  int*   t0buf = (int*)alloc((size_t)4 * 4 * 1024 * 16 * NH);
  int*   t1buf = (int*)alloc((size_t)4 * 4 * 4096 * 8 * NH);

  const float* fin[9]  = {q0, q1, q2, k0, k1, k2, v0, v1, v2};
  unsigned short* fo[9] = {pq0, pq1, pq2, pk0, pk1, pk2, pv0, pv1, pv2};
  const int Ss[9] = {S0, S1, S2, S0, S1, S2, S0, S1, S2};
  for (int i = 0; i < 9; ++i)
    pack_kernel<<<dim3(Ss[i] / 32, NH, 4), dim3(256), 0, stream>>>(fin[i], fo[i], Ss[i]);

  coarse_kernel<<<dim3(64 / CW, NH, 4), dim3(32 * CW), 0, stream>>>(
      pq2, pk2, pv2, m0buf, t0buf);

  fine_kernel<64, 16, true><<<dim3(1024, NH, 4), dim3(32), 0, stream>>>(
      pq1, pk1, pv1, t0buf, m1buf, t1buf, 64);

  fine_kernel<32, 8, false><<<dim3(4096, NH, 4), dim3(32), 0, stream>>>(
      pq0, pk0, pv0, t1buf, m2buf, (int*)nullptr, 128);

  combine_kernel<<<dim3((4u * 16384 * NH * D) / 256, 1, 1), dim3(256), 0, stream>>>(
      m0buf, m1buf, m2buf, wt, (float*)d_out);
}
